// MPNN_sparse_5566277616082
// MI455X (gfx1250) — matrix operand order repaired and checked by reading, not compiled, not run on hardware
//
#include <hip/hip_runtime.h>

// ---------------------------------------------------------------------------
// Types for CDNA5 WMMA (wave32, 16x16x32 bf16 -> f32) and TDM descriptors
// ---------------------------------------------------------------------------
typedef __bf16 v16bf __attribute__((ext_vector_type(16)));
typedef __bf16 v8bf  __attribute__((ext_vector_type(8)));
typedef float  v8f   __attribute__((ext_vector_type(8)));
typedef unsigned int u32x4 __attribute__((ext_vector_type(4)));
typedef int          i32x8 __attribute__((ext_vector_type(8)));
typedef int          i32x4 __attribute__((ext_vector_type(4)));

#define D_IN   128
#define D_H    512
#define D_MSG  128
#define K1     256          // 2*D_IN (message) / D_IN+D_MSG (update)
#define ASTR   264          // A-tile LDS row stride in bf16 (256 + 8 pad = 528B)
#define HSTR   520          // H-tile LDS row stride in bf16 (512 + 8 pad = 1040B)
#define ROWS   32           // edges/nodes per workgroup (2 row tiles of 16)

#if __has_builtin(__builtin_amdgcn_tensor_load_to_lds)
#define HAVE_TDM 1
#endif

__device__ __forceinline__ unsigned f32_to_bf16(float f) {
    union { float f; unsigned u; } c; c.f = f;
    unsigned u = c.u;
    u += 0x7FFFu + ((u >> 16) & 1u);          // round to nearest even
    return u >> 16;
}

// A fragment: 16x32 bf16. lane<16 : row=lane,    K = {0..15}
//             lane>=16: row=lane-16, K = {16..31}
__device__ __forceinline__ v16bf load_a_frag(const unsigned short* rowk0, int halfsel) {
    union { v16bf v; v8bf h[2]; } u;
    u.h[0] = *(const v8bf*)(rowk0 + halfsel * 16);
    u.h[1] = *(const v8bf*)(rowk0 + halfsel * 16 + 8);
    return u.v;
}

// B fragment: 32x16 bf16 from W^T[fout][fin]; lane = col n0+lane%16,
// 16 consecutive K starting at k0 + (lane>=16)*16  -> two b128 loads.
__device__ __forceinline__ v16bf load_b_frag(const unsigned short* wk) {
    union { v16bf v; v8bf h[2]; } u;
    u.h[0] = *(const v8bf*)(wk);
    u.h[1] = *(const v8bf*)(wk + 8);
    return u.v;
}

#if defined(HAVE_TDM)
// TDM gather-mode load: 16 rows of 128 bf16 gathered by 16-bit row indices
// from xb[N][128] into LDS at lds_byte_addr with 528-byte row pitch
// (pad_enable: 64 DWORDs data + 68 DWORDs pad).
__device__ __forceinline__ void tdm_gather_rows16(
    const unsigned short* xb, unsigned lds_byte_addr, int N,
    const long long* idxp /* 16 consecutive int64 row indices */)
{
    unsigned long long ga = (unsigned long long)(size_t)xb;
    u32x4 g0;
    g0[0] = 1u | (1u << 31);                                  // count=1, gather_mode, 16-bit idx
    g0[1] = lds_byte_addr;                                    // LDS dest (bytes)
    g0[2] = (unsigned)ga;                                     // global_addr[31:0]
    g0[3] = (unsigned)((ga >> 32) & 0x01FFFFFFull) | (2u << 30); // addr[56:32] | type=2

    i32x8 g1;
    g1[0] = (int)((1u << 16)      // data_size = 2 bytes
                | (1u << 20)      // pad_enable
                | (5u << 22)      // pad_interval: 64 DWORDs (256B of row data)
                | (67u << 25));   // pad_amount:  68 DWORDs (272B -> 528B pitch)
    g1[1] = (int)(128u << 16);                                 // tensor_dim0 = 128
    g1[2] = (int)(((unsigned)N & 0xFFFFu) << 16);              // tensor_dim1 lo16
    g1[3] = (int)(((unsigned)N >> 16) | (128u << 16));         // tensor_dim1 hi | tile_dim0=128
    g1[4] = 16;                                                // tile_dim1 = #valid indices
    g1[5] = 128;                                               // tensor_dim0_stride = 128 elems
    g1[6] = 0;
    g1[7] = 0;

    i32x4 g2, g3;
    #pragma unroll
    for (int k = 0; k < 4; ++k) {
        unsigned a = (unsigned)(unsigned short)idxp[2 * k];
        unsigned b = (unsigned)(unsigned short)idxp[2 * k + 1];
        g2[k] = (int)(a | (b << 16));
        unsigned c = (unsigned)(unsigned short)idxp[8 + 2 * k];
        unsigned d = (unsigned)(unsigned short)idxp[8 + 2 * k + 1];
        g3[k] = (int)(c | (d << 16));
    }
    i32x8 g4 = { 0, 0, 0, 0, 0, 0, 0, 0 };   // unused extra group (6-arg toolchain form)
    __builtin_amdgcn_tensor_load_to_lds(g0, g1, g2, g3, g4, 0);
}
#endif

// ---------------------------------------------------------------------------
// Prep kernels
// ---------------------------------------------------------------------------
__global__ void zero_f32_kernel(float* __restrict__ p, int n) {
    int i = blockIdx.x * blockDim.x + threadIdx.x;
    if (i < n) p[i] = 0.0f;
}

// f32 -> bf16, 4 elements per thread
__global__ void convert_bf16_kernel(const float* __restrict__ in,
                                    unsigned short* __restrict__ out, int n4) {
    int i = blockIdx.x * blockDim.x + threadIdx.x;
    if (i >= n4) return;
    float4 v = ((const float4*)in)[i];
    unsigned lo = f32_to_bf16(v.x) | (f32_to_bf16(v.y) << 16);
    unsigned hi = f32_to_bf16(v.z) | (f32_to_bf16(v.w) << 16);
    ((uint2*)out)[i] = make_uint2(lo, hi);
}

// W[rows][cols] f32 -> WT[cols][rows] bf16
__global__ void transpose_bf16_kernel(const float* __restrict__ W,
                                      unsigned short* __restrict__ WT,
                                      int rows, int cols) {
    int i = blockIdx.x * blockDim.x + threadIdx.x;
    if (i >= rows * cols) return;
    int r = i / cols;
    int c = i - r * cols;
    WT[c * rows + r] = (unsigned short)f32_to_bf16(W[i]);
}

// ---------------------------------------------------------------------------
// Shared GEMM1 (+bias, ReLU): A[32x256] -> H[32x512], one wave owns 128 cols,
// each B fragment reused for both 16-row tiles.
// ---------------------------------------------------------------------------
__device__ __forceinline__ void mlp_layer1(
    const unsigned short* Atile, unsigned short* Htile,
    const unsigned short* W1T, const float* __restrict__ b1,
    int wave, int lane)
{
    const int col16 = lane & 15;
    const int half  = lane >> 4;
    const unsigned short* a0 = Atile + col16 * ASTR;
    const unsigned short* a1 = Atile + (16 + col16) * ASTR;
    #pragma unroll
    for (int t = 0; t < 8; ++t) {
        const int n0 = wave * 128 + t * 16;
        const float bv = b1[n0 + col16];
        v8f acc0 = { bv, bv, bv, bv, bv, bv, bv, bv };
        v8f acc1 = acc0;
        const unsigned short* wb = W1T + (size_t)(n0 + col16) * K1 + half * 16;
        #pragma unroll
        for (int k = 0; k < K1; k += 32) {
            v16bf b = load_b_frag(wb + k);
            acc0 = __builtin_amdgcn_wmma_f32_16x16x32_bf16(
                       false, load_a_frag(a0 + k, half), false, b, (short)0, acc0, false, false);
            acc1 = __builtin_amdgcn_wmma_f32_16x16x32_bf16(
                       false, load_a_frag(a1 + k, half), false, b, (short)0, acc1, false, false);
        }
        #pragma unroll
        for (int r = 0; r < 8; ++r) {
            const int m = r + half * 8;
            float v0 = acc0[r] > 0.0f ? acc0[r] : 0.0f;
            float v1 = acc1[r] > 0.0f ? acc1[r] : 0.0f;
            Htile[m * HSTR + n0 + col16]        = (unsigned short)f32_to_bf16(v0);
            Htile[(16 + m) * HSTR + n0 + col16] = (unsigned short)f32_to_bf16(v1);
        }
    }
}

// ---------------------------------------------------------------------------
// Message kernel: 32 edges per block, 128 threads (4 wave32)
// ---------------------------------------------------------------------------
__global__ __launch_bounds__(128) void mpnn_msg_kernel(
    const unsigned short* __restrict__ xb,      // [N][128] bf16
    const long long* __restrict__ edge_index,   // [2][E], row0=src j, row1=dst i
    const unsigned short* __restrict__ W1T,     // [512][256] bf16
    const float* __restrict__ b1,
    const unsigned short* __restrict__ W2T,     // [128][512] bf16
    const float* __restrict__ b2,
    float* __restrict__ agg,                    // [N][128] f32
    long long E, int N)
{
    __shared__ unsigned short Atile[ROWS * ASTR];
    __shared__ unsigned short Htile[ROWS * HSTR];
    __shared__ int idxI[ROWS];

    const int tid   = threadIdx.x;
    const int lane  = tid & 31;
    const int wave  = tid >> 5;
    const int col16 = lane & 15;
    const int half  = lane >> 4;
    const long long ebase = (long long)blockIdx.x * ROWS;

    // ---- phase 1: gather [x_i | x_j] (bf16) into A tile ----
#if defined(HAVE_TDM)
    if (wave == 0) {
        const unsigned aBase = (unsigned)(size_t)&Atile[0];
        // x_i -> cols 0..127, x_j -> cols 128..255; two 16-row tiles each
        tdm_gather_rows16(xb, aBase,                       N, edge_index + E + ebase);
        tdm_gather_rows16(xb, aBase + 16 * 2 * ASTR,       N, edge_index + E + ebase + 16);
        tdm_gather_rows16(xb, aBase + 256,                 N, edge_index + ebase);
        tdm_gather_rows16(xb, aBase + 16 * 2 * ASTR + 256, N, edge_index + ebase + 16);
        __builtin_amdgcn_s_wait_tensorcnt(0);
    }
#else
    {
        const int row = tid >> 2;               // 0..31
        const int q   = tid & 3;                // quarter of the 512B row
        const long long e = ebase + row;
        const long long node = (q < 2) ? edge_index[E + e] : edge_index[e];
        const unsigned short* src = xb + node * D_IN + (q & 1) * 64;
        unsigned short* dst = Atile + row * ASTR + q * 64;
        #pragma unroll
        for (int c = 0; c < 64; c += 8)
            *(uint4*)(dst + c) = *(const uint4*)(src + c);
    }
#endif
    if (tid < ROWS) idxI[tid] = (int)edge_index[E + ebase + tid];
    __syncthreads();

    // ---- phase 2: H = relu(A @ W1m + b1m) ----
    mlp_layer1(Atile, Htile, W1T, b1, wave, lane);
    __syncthreads();

    // ---- phase 3: msgs = H @ W2m + b2m; scatter-add into agg[idx_i] ----
    const unsigned short* h0 = Htile + col16 * HSTR;
    const unsigned short* h1 = Htile + (16 + col16) * HSTR;
    #pragma unroll
    for (int t = 0; t < 2; ++t) {
        const int n0 = wave * 32 + t * 16;
        const float bv = b2[n0 + col16];
        v8f acc0 = { bv, bv, bv, bv, bv, bv, bv, bv };
        v8f acc1 = acc0;
        const unsigned short* wb = W2T + (size_t)(n0 + col16) * D_H + half * 16;
        #pragma unroll
        for (int k = 0; k < D_H; k += 32) {
            v16bf b = load_b_frag(wb + k);
            acc0 = __builtin_amdgcn_wmma_f32_16x16x32_bf16(
                       false, load_a_frag(h0 + k, half), false, b, (short)0, acc0, false, false);
            acc1 = __builtin_amdgcn_wmma_f32_16x16x32_bf16(
                       false, load_a_frag(h1 + k, half), false, b, (short)0, acc1, false, false);
        }
        #pragma unroll
        for (int r = 0; r < 8; ++r) {
            const int m = r + half * 8;
            atomicAdd(agg + (size_t)idxI[m] * D_MSG + n0 + col16, acc0[r]);
            atomicAdd(agg + (size_t)idxI[16 + m] * D_MSG + n0 + col16, acc1[r]);
        }
    }
}

// ---------------------------------------------------------------------------
// Update kernel: 32 nodes per block (last block may be 16-row tail)
// ---------------------------------------------------------------------------
__global__ __launch_bounds__(128) void mpnn_upd_kernel(
    const unsigned short* __restrict__ xb,      // [N][128] bf16
    const unsigned short* __restrict__ aggb,    // [N][128] bf16
    const unsigned short* __restrict__ W1T,     // [512][256] bf16 (W1u^T)
    const float* __restrict__ b1,
    const unsigned short* __restrict__ W2T,     // [128][512] bf16 (W2u^T)
    const float* __restrict__ b2,
    float* __restrict__ out,                    // [N][128] f32
    int N)
{
    __shared__ unsigned short Atile[ROWS * ASTR];
    __shared__ unsigned short Htile[ROWS * HSTR];

    const int tid   = threadIdx.x;
    const int lane  = tid & 31;
    const int wave  = tid >> 5;
    const int col16 = lane & 15;
    const int half  = lane >> 4;
    const long long nbase = (long long)blockIdx.x * ROWS;

    // ---- phase 1: gather [x | agg] (bf16, contiguous rows, clamped) ----
    {
        const int row = tid >> 2;
        const int q   = tid & 3;
        long long node = nbase + row;
        if (node >= N) node = N - 1;            // tail clamp (results unused)
        const unsigned short* src = ((q < 2) ? xb : aggb) + node * D_IN + (q & 1) * 64;
        unsigned short* dst = Atile + row * ASTR + q * 64;
        #pragma unroll
        for (int c = 0; c < 64; c += 8)
            *(uint4*)(dst + c) = *(const uint4*)(src + c);
    }
    __syncthreads();

    // ---- phase 2: H = relu(A @ W1u + b1u) ----
    mlp_layer1(Atile, Htile, W1T, b1, wave, lane);
    __syncthreads();

    // ---- phase 3: out = H @ W2u + b2u ----
    const unsigned short* h0 = Htile + col16 * HSTR;
    const unsigned short* h1 = Htile + (16 + col16) * HSTR;
    #pragma unroll
    for (int t = 0; t < 2; ++t) {
        const int n0 = wave * 32 + t * 16;
        const float bv = b2[n0 + col16];
        v8f acc0 = { bv, bv, bv, bv, bv, bv, bv, bv };
        v8f acc1 = acc0;
        const unsigned short* wb = W2T + (size_t)(n0 + col16) * D_H + half * 16;
        #pragma unroll
        for (int k = 0; k < D_H; k += 32) {
            v16bf b = load_b_frag(wb + k);
            acc0 = __builtin_amdgcn_wmma_f32_16x16x32_bf16(
                       false, load_a_frag(h0 + k, half), false, b, (short)0, acc0, false, false);
            acc1 = __builtin_amdgcn_wmma_f32_16x16x32_bf16(
                       false, load_a_frag(h1 + k, half), false, b, (short)0, acc1, false, false);
        }
        #pragma unroll
        for (int r = 0; r < 8; ++r) {
            const int m = r + half * 8;
            const long long n0r = nbase + m;
            const long long n1r = nbase + 16 + m;
            if (n0r < N) out[n0r * D_MSG + n0 + col16] = acc0[r];
            if (n1r < N) out[n1r * D_MSG + n0 + col16] = acc1[r];
        }
    }
}

// ---------------------------------------------------------------------------
// Launch
// ---------------------------------------------------------------------------
extern "C" void kernel_launch(void* const* d_in, const int* in_sizes, int n_in,
                              void* d_out, int out_size, void* d_ws, size_t ws_size,
                              hipStream_t stream) {
    const float*     x          = (const float*)d_in[0];
    // d_in[1] = degrees (unused by the reference)
    const long long* edge_index = (const long long*)d_in[2];   // int64 [2][E]
    const float*     W1m = (const float*)d_in[3];
    const float*     b1m = (const float*)d_in[4];
    const float*     W2m = (const float*)d_in[5];
    const float*     b2m = (const float*)d_in[6];
    const float*     W1u = (const float*)d_in[7];
    const float*     b1u = (const float*)d_in[8];
    const float*     W2u = (const float*)d_in[9];
    const float*     b2u = (const float*)d_in[10];
    float*           out = (float*)d_out;

    const int N = in_sizes[0] / D_IN;       // 50000
    const long long E = in_sizes[2] / 2;    // 800000

    // workspace layout
    char* ws = (char*)d_ws;
    size_t off = 0;
    float*          agg   = (float*)(ws + off);          off += (size_t)N * D_MSG * 4;
    unsigned short* xb16  = (unsigned short*)(ws + off); off += (size_t)N * D_IN  * 2;
    unsigned short* aggb  = (unsigned short*)(ws + off); off += (size_t)N * D_MSG * 2;
    unsigned short* W1mT  = (unsigned short*)(ws + off); off += (size_t)K1  * D_H   * 2;
    unsigned short* W2mT  = (unsigned short*)(ws + off); off += (size_t)D_H * D_MSG * 2;
    unsigned short* W1uT  = (unsigned short*)(ws + off); off += (size_t)K1  * D_H   * 2;
    unsigned short* W2uT  = (unsigned short*)(ws + off);

    const int nFeat = N * D_MSG;            // == N * D_IN

    // 1) zero agg (every call — harness does not re-zero scratch)
    zero_f32_kernel<<<(nFeat + 255) / 256, 256, 0, stream>>>(agg, nFeat);

    // 2) bf16 copies / transposed bf16 weights
    convert_bf16_kernel<<<(nFeat / 4 + 255) / 256, 256, 0, stream>>>(x, xb16, nFeat / 4);
    transpose_bf16_kernel<<<(K1 * D_H + 255) / 256, 256, 0, stream>>>(W1m, W1mT, K1, D_H);
    transpose_bf16_kernel<<<(D_H * D_MSG + 255) / 256, 256, 0, stream>>>(W2m, W2mT, D_H, D_MSG);
    transpose_bf16_kernel<<<(K1 * D_H + 255) / 256, 256, 0, stream>>>(W1u, W1uT, K1, D_H);
    transpose_bf16_kernel<<<(D_H * D_MSG + 255) / 256, 256, 0, stream>>>(W2u, W2uT, D_H, D_MSG);

    // 3) message MLP + scatter (E/32 = 25000 tiles)
    mpnn_msg_kernel<<<(int)(E / ROWS), 128, 0, stream>>>(
        xb16, edge_index, W1mT, b1m, W2mT, b2m, agg, E, N);

    // 4) agg -> bf16, then update MLP
    convert_bf16_kernel<<<(nFeat / 4 + 255) / 256, 256, 0, stream>>>(agg, aggb, nFeat / 4);
    mpnn_upd_kernel<<<(N + ROWS - 1) / ROWS, 128, 0, stream>>>(
        xb16, aggb, W1uT, b1u, W2uT, b2u, out, N);
}
